// Qwen3InfAttention_5248450036536
// MI455X (gfx1250) — compile-verified, matrix operand order checked by reading, not run on hardware
//
#include <hip/hip_runtime.h>
#include <hip/hip_bf16.h>

// ---------- problem constants ----------
#define B_   2
#define S_   2048
#define HID_ 2048
#define NH_  16
#define NKV_ 8
#define HD_  128
#define QKV_N 4096              // 16*128 + 8*128 + 8*128
#define M_ROWS (B_*S_)          // 4096

typedef unsigned short u16;
typedef __bf16 bf16_t;
typedef bf16_t bf16x16 __attribute__((ext_vector_type(16)));
typedef float  f32x8   __attribute__((ext_vector_type(8)));
typedef unsigned v4u   __attribute__((ext_vector_type(4)));
typedef unsigned v8u   __attribute__((ext_vector_type(8)));

// ---------- helpers ----------
__device__ __forceinline__ u16 f2bf(float x) {
    union { float f; unsigned u; } c; c.f = x;
    unsigned r = c.u + 0x7fffu + ((c.u >> 16) & 1u);   // RNE
    return (u16)(r >> 16);
}

// 16x32 bf16 WMMA fragment: lane supplies its row base; halves are
// rowbase[k0..k0+7] and rowbase[k0+16..k0+23], k0 = (lane>>4)*8 (+tile offset).
__device__ __forceinline__ bf16x16 ld_frag(const u16* rowbase, int k0) {
    union { uint4 u[2]; bf16x16 v; } f;
    f.u[0] = *(const uint4*)(rowbase + k0);
    f.u[1] = *(const uint4*)(rowbase + k0 + 16);
    return f.v;
}

// ---------- kernel 0: f32 -> bf16 ----------
__global__ void cvt_f32_bf16(const float* __restrict__ in, u16* __restrict__ out, int n) {
    int i = blockIdx.x * blockDim.x + threadIdx.x;
    if (i < n) out[i] = f2bf(in[i]);
}

// ---------- kernel 1/4: WMMA GEMM  C[M][N] = A[M][K] * W[N][K]^T ----------
// Block tile 128(M) x 128(N), 256 threads = 8 waves (4x2), wave tile 32x64.
// Double-buffered LDS staging: A tile via one TDM tensor_load_to_lds descriptor
// per stage (wave 0), B tile via per-lane global_load_async_to_lds_b128.
#define BM 128
#define BN 128
#define BK 32
__global__ __launch_bounds__(256) void gemm_bf16_nt(const u16* __restrict__ A,
                                                    const u16* __restrict__ W,
                                                    float* __restrict__ C,
                                                    int M, int N, int K) {
    __shared__ __align__(16) u16 sA[2][BM * BK];   // 2 x 8 KiB
    __shared__ __align__(16) u16 sB[2][BN * BK];   // 2 x 8 KiB

    const int tid  = threadIdx.x;
    const int lane = tid & 31;
    const int wave = tid >> 5;
    const int wm   = wave >> 1;     // 0..3  (M direction)
    const int wn   = wave & 1;      // 0..1  (N direction, 64 cols each)
    const int mBlk = blockIdx.y * BM;
    const int nBlk = blockIdx.x * BN;
    const int kfo  = (lane >> 4) << 3;

    f32x8 acc[2][4] = {};

    // ---- tile staging (producer side) ----
    auto stage = [&](int buf, int k0) {
        // B tile: 128 rows x 32 halves, per-lane async copies (2 x B128/thread)
        const int row = tid >> 1;
        const int col = (tid & 1) * 16;
        unsigned long long gb =
            (unsigned long long)(uintptr_t)(W + (size_t)(nBlk + row) * K + k0 + col);
        unsigned lb = (unsigned)(uintptr_t)&sB[buf][row * BK + col];
        asm volatile("global_load_async_to_lds_b128 %0, %1, off"
                     :: "v"(lb), "v"(gb) : "memory");
        asm volatile("global_load_async_to_lds_b128 %0, %1, off"
                     :: "v"(lb + 16u), "v"(gb + 16ull) : "memory");

        // A tile: one 2D TDM descriptor (32 halves x 128 rows, stride K)
        if (wave == 0) {
            unsigned lds = (unsigned)(uintptr_t)&sA[buf][0];
            unsigned long long ga =
                (unsigned long long)(uintptr_t)(A + (size_t)mBlk * K + k0);
            v4u g0;
            g0[0] = 1u;                                        // count=1, user mode
            g0[1] = lds;                                       // lds_addr [63:32]
            g0[2] = (unsigned)ga;                              // global_addr lo
            g0[3] = (unsigned)((ga >> 32) & 0x01FFFFFFull)     // global_addr [56:32]
                    | (2u << 30);                              // type=2 (image)
            v8u g1;
            g1[0] = 0x00010000u;          // wg_mask=0, data_size=1 (2 bytes)
            g1[1] = ((unsigned)BK) << 16; // tensor_dim0[15:0] = 32
            g1[2] = ((unsigned)BM) << 16; // tensor_dim1[15:0] = 128
            g1[3] = ((unsigned)BK) << 16; // tile_dim0 = 32
            g1[4] = (unsigned)BM;         // tile_dim1 = 128, tile_dim2 = 0
            g1[5] = (unsigned)K;          // tensor_dim0_stride[31:0] = K
            g1[6] = 0u;
            g1[7] = 0u;
            asm volatile("tensor_load_to_lds %0, %1" :: "s"(g0), "s"(g1) : "memory");
        }
    };

    stage(0, 0);
    const int nIter = K / BK;
    for (int it = 0; it < nIter; ++it) {
        const int cur = it & 1;
        // consume gate: this wave's DMAs done, then all waves' DMAs done
        asm volatile("s_wait_asynccnt 0x0" ::: "memory");
        __builtin_amdgcn_s_wait_tensorcnt(0);
        __syncthreads();
        // overlap: kick next tile's DMA before computing on current tile
        if (it + 1 < nIter) stage(1 - cur, (it + 1) * BK);

        const u16* pA = &sA[cur][0];
        const u16* pB = &sB[cur][0];
        bf16x16 bfr[4];
        #pragma unroll
        for (int ns = 0; ns < 4; ++ns)
            bfr[ns] = ld_frag(pB + (wn * 64 + ns * 16 + (lane & 15)) * BK, kfo);
        #pragma unroll
        for (int ms = 0; ms < 2; ++ms) {
            bf16x16 af = ld_frag(pA + (wm * 32 + ms * 16 + (lane & 15)) * BK, kfo);
            #pragma unroll
            for (int ns = 0; ns < 4; ++ns)
                acc[ms][ns] = __builtin_amdgcn_wmma_f32_16x16x32_bf16(
                    false, af, false, bfr[ns], (short)0, acc[ms][ns], false, false);
        }
        __syncthreads();   // all waves done reading 'cur' before it is re-staged
    }

    const int rbase = (lane >> 4) << 3;   // C layout: M = r + 8*(lane/16)
    #pragma unroll
    for (int ms = 0; ms < 2; ++ms)
        #pragma unroll
        for (int ns = 0; ns < 4; ++ns)
            #pragma unroll
            for (int r = 0; r < 8; ++r) {
                int row = mBlk + wm * 32 + ms * 16 + r + rbase;
                int col = nBlk + wn * 64 + ns * 16 + (lane & 15);
                C[(size_t)row * N + col] = acc[ms][ns][r];
            }
}

// ---------- kernel 2: RMSNorm + RoPE + layout transform ----------
__global__ __launch_bounds__(128) void norm_rope(const float* __restrict__ qkv,
                                                 const float* __restrict__ cosb,
                                                 const float* __restrict__ sinb,
                                                 const float* __restrict__ qw,
                                                 const float* __restrict__ kw,
                                                 u16* __restrict__ qout,
                                                 u16* __restrict__ kout,
                                                 u16* __restrict__ vtout) {
    int bid  = blockIdx.x;
    int slot = bid & 31;
    int bs   = bid >> 5;                 // b*S + s
    int s    = bs & (S_ - 1);
    int b    = bs >> 11;
    int d    = threadIdx.x;

    float x = qkv[(size_t)bs * QKV_N + slot * HD_ + d];

    if (slot < NH_ + NKV_) {             // Q or K: RMSNorm + RoPE
        __shared__ float red[HD_];
        __shared__ float sh[HD_];
        red[d] = x * x;
        __syncthreads();
        #pragma unroll
        for (int off = 64; off > 0; off >>= 1) {
            if (d < off) red[d] += red[d + off];
            __syncthreads();
        }
        float inv = rsqrtf(red[0] * (1.0f / HD_) + 1e-6f);
        const float* w = (slot < NH_) ? qw : kw;
        float xn = x * inv * w[d];
        sh[d] = xn;
        __syncthreads();
        float rot = (d < 64) ? -sh[d + 64] : sh[d - 64];
        float c  = cosb[(size_t)bs * HD_ + d];
        float sn = sinb[(size_t)bs * HD_ + d];
        float o = xn * c + rot * sn;
        if (slot < NH_) {
            qout[(((size_t)b * NH_ + slot) * S_ + s) * HD_ + d] = f2bf(o);
        } else {
            kout[(((size_t)b * NKV_ + (slot - NH_)) * S_ + s) * HD_ + d] = f2bf(o);
        }
    } else {                             // V: transpose to [b][kv][d][s]
        int h = slot - (NH_ + NKV_);
        vtout[(((size_t)b * NKV_ + h) * HD_ + d) * S_ + s] = f2bf(x);
    }
}

// ---------- kernel 3: GQA flash attention ----------
__global__ __launch_bounds__(128) void flash_attn(const u16* __restrict__ Q,
                                                  const u16* __restrict__ Kmat,
                                                  const u16* __restrict__ Vt,
                                                  u16* __restrict__ attn) {
    int bid  = blockIdx.x;
    int qt   = bid & 31;                 // S/64 q-tiles
    int h    = (bid >> 5) & (NH_ - 1);
    int b    = bid >> 9;
    int wave = threadIdx.x >> 5, lane = threadIdx.x & 31;
    int q0   = qt * 64 + wave * 16;
    int hk   = h >> 1;                   // GROUPS = 2
    const int kfo = (lane >> 4) << 3;

    const u16* qbase = Q + (((size_t)b * NH_ + h) * S_ + q0 + (lane & 15)) * HD_;
    const u16* kbase = Kmat + (((size_t)b * NKV_ + hk) * S_) * HD_;
    const u16* vbase = Vt   + (((size_t)b * NKV_ + hk) * HD_) * S_;

    bf16x16 qf[4];
    #pragma unroll
    for (int kk = 0; kk < 4; ++kk) qf[kk] = ld_frag(qbase, kk * 32 + kfo);

    __shared__ __align__(16) u16 sP[4][16 * 32];

    float m[8], l[8], alpha[8];
    #pragma unroll
    for (int r = 0; r < 8; ++r) { m[r] = -1e30f; l[r] = 0.f; }
    f32x8 o[8] = {};

    const float scale = 0.08838834764831845f;  // 128^-0.5

    for (int j0 = 0; j0 < S_; j0 += 32) {
        f32x8 s0 = {}, s1 = {};
        #pragma unroll
        for (int kk = 0; kk < 4; ++kk) {
            bf16x16 kb0 = ld_frag(kbase + (size_t)(j0 +      (lane & 15)) * HD_, kk * 32 + kfo);
            bf16x16 kb1 = ld_frag(kbase + (size_t)(j0 + 16 + (lane & 15)) * HD_, kk * 32 + kfo);
            s0 = __builtin_amdgcn_wmma_f32_16x16x32_bf16(false, qf[kk], false, kb0,
                     (short)0, s0, false, false);
            s1 = __builtin_amdgcn_wmma_f32_16x16x32_bf16(false, qf[kk], false, kb1,
                     (short)0, s1, false, false);
        }
        // online softmax (rows live in 16-lane halves of the C layout)
        #pragma unroll
        for (int r = 0; r < 8; ++r) {
            float a = s0[r] * scale, c = s1[r] * scale;
            float mx = fmaxf(a, c);
            #pragma unroll
            for (int off = 1; off < 16; off <<= 1) mx = fmaxf(mx, __shfl_xor(mx, off, 32));
            float mn = fmaxf(m[r], mx);
            alpha[r] = __expf(m[r] - mn);
            float p0 = __expf(a - mn), p1 = __expf(c - mn);
            float rs = p0 + p1;
            #pragma unroll
            for (int off = 1; off < 16; off <<= 1) rs += __shfl_xor(rs, off, 32);
            l[r] = l[r] * alpha[r] + rs;
            m[r] = mn;
            int prow = r + kfo;          // C-layout -> A-layout via LDS
            sP[wave][prow * 32 +      (lane & 15)] = f2bf(p0);
            sP[wave][prow * 32 + 16 + (lane & 15)] = f2bf(p1);
        }
        #pragma unroll
        for (int n = 0; n < 8; ++n)
            #pragma unroll
            for (int r = 0; r < 8; ++r) o[n][r] *= alpha[r];

        // P (16x32 A-frag from LDS) x V (32x128 from transposed V)
        bf16x16 pa = ld_frag(&sP[wave][(lane & 15) * 32], kfo);
        #pragma unroll
        for (int n = 0; n < 8; ++n) {
            bf16x16 vb = ld_frag(vbase + (size_t)(n * 16 + (lane & 15)) * S_ + j0, kfo);
            o[n] = __builtin_amdgcn_wmma_f32_16x16x32_bf16(false, pa, false, vb,
                       (short)0, o[n], false, false);
        }
    }

    #pragma unroll
    for (int n = 0; n < 8; ++n)
        #pragma unroll
        for (int r = 0; r < 8; ++r) {
            int row = q0 + r + kfo;
            float val = o[n][r] / l[r];
            attn[((size_t)(b * S_ + row)) * (NH_ * HD_) + h * HD_ + n * 16 + (lane & 15)] = f2bf(val);
        }
}

// ---------- host side ----------
extern "C" void kernel_launch(void* const* d_in, const int* in_sizes, int n_in,
                              void* d_out, int out_size, void* d_ws, size_t ws_size,
                              hipStream_t stream) {
    const float* hidden = (const float*)d_in[0];
    const float* cosb   = (const float*)d_in[1];
    const float* sinb   = (const float*)d_in[2];
    const float* Wq     = (const float*)d_in[3];
    const float* Wk     = (const float*)d_in[4];
    const float* Wv     = (const float*)d_in[5];
    const float* Wo     = (const float*)d_in[6];
    const float* qw     = (const float*)d_in[7];
    const float* kw     = (const float*)d_in[8];
    float* out = (float*)d_out;

    // workspace layout: ~152 MiB
    char* p = (char*)d_ws;
    u16*   hid_bf  = (u16*)p;  p += (size_t)M_ROWS * HID_ * 2;
    u16*   wqkv_bf = (u16*)p;  p += (size_t)QKV_N * HID_ * 2;
    u16*   wo_bf   = (u16*)p;  p += (size_t)HID_ * (NH_*HD_) * 2;
    float* qkv     = (float*)p; p += (size_t)M_ROWS * QKV_N * 4;
    u16*   qb      = (u16*)p;  p += (size_t)B_ * NH_  * S_ * HD_ * 2;
    u16*   kb      = (u16*)p;  p += (size_t)B_ * NKV_ * S_ * HD_ * 2;
    u16*   vtb     = (u16*)p;  p += (size_t)B_ * NKV_ * HD_ * S_ * 2;
    u16*   attnb   = (u16*)p;  p += (size_t)M_ROWS * (NH_*HD_) * 2;

    // 0) bf16 conversions
    {
        int n = M_ROWS * HID_;
        cvt_f32_bf16<<<(n + 255) / 256, 256, 0, stream>>>(hidden, hid_bf, n);
        n = (NH_ * HD_) * HID_;
        cvt_f32_bf16<<<(n + 255) / 256, 256, 0, stream>>>(Wq, wqkv_bf, n);
        int nk = (NKV_ * HD_) * HID_;
        cvt_f32_bf16<<<(nk + 255) / 256, 256, 0, stream>>>(Wk, wqkv_bf + (size_t)(NH_*HD_) * HID_, nk);
        cvt_f32_bf16<<<(nk + 255) / 256, 256, 0, stream>>>(Wv, wqkv_bf + (size_t)(NH_*HD_ + NKV_*HD_) * HID_, nk);
        n = HID_ * (NH_ * HD_);
        cvt_f32_bf16<<<(n + 255) / 256, 256, 0, stream>>>(Wo, wo_bf, n);
    }

    // 1) fused QKV projection -> [4096 x 4096] f32
    gemm_bf16_nt<<<dim3(QKV_N / BN, M_ROWS / BM), 256, 0, stream>>>(
        hid_bf, wqkv_bf, qkv, M_ROWS, QKV_N, HID_);

    // 2) RMSNorm + RoPE + transposes
    norm_rope<<<B_ * S_ * 32, HD_, 0, stream>>>(qkv, cosb, sinb, qw, kw, qb, kb, vtb);

    // 3) GQA flash attention
    flash_attn<<<B_ * NH_ * (S_ / 64), 128, 0, stream>>>(qb, kb, vtb, attnb);

    // 4) output projection -> d_out (f32)
    gemm_bf16_nt<<<dim3(HID_ / BN, M_ROWS / BM), 256, 0, stream>>>(
        attnb, wo_bf, out, M_ROWS, HID_, NH_ * HD_);
}
